// MultiHeadAttention_37890201485506
// MI455X (gfx1250) — matrix operand order repaired and checked by reading, not compiled, not run on hardware
//
#include <hip/hip_runtime.h>
#include <hip/hip_bf16.h>

// Problem constants (match reference)
#define BB 8
#define NN 1024
#define DDIM 512
#define HH 8
#define HD 64
#define MM (BB * NN)          // 8192 rows
#define ATT_SCALE 0.125f      // HD^-0.5
#define MASKNEG (-1e30f)
#define KSTEPS (DDIM / 32)    // 16 k-iterations per GEMM

typedef _Float16 v16h __attribute__((ext_vector_type(16)));
typedef _Float16 half8 __attribute__((ext_vector_type(8)));
typedef _Float16 half4 __attribute__((ext_vector_type(4)));
typedef float    v8f  __attribute__((ext_vector_type(8)));

// ---------------------------------------------------------------------------
// Async global->LDS copy (CDNA5 GLOBAL_LOAD_ASYNC_TO_LDS_B128, ASYNCcnt).
// Probe-learned signature: (int4 AS1* gsrc, int4 AS3* ldsdst, imm off, imm cpol)
// ---------------------------------------------------------------------------
#if __has_builtin(__builtin_amdgcn_global_load_async_to_lds_b128)
#define HAVE_ASYNC_LDS 1
#else
#define HAVE_ASYNC_LDS 0
#warning "global_load_async_to_lds_b128 builtin not available; synchronous LDS staging fallback"
#endif

typedef int v4i_vs __attribute__((vector_size(16)));
typedef __attribute__((address_space(1))) void as1_void;
typedef __attribute__((address_space(3))) void as3_void;
typedef __attribute__((address_space(1))) v4i_vs as1_v4i;
typedef __attribute__((address_space(3))) v4i_vs as3_v4i;

static __device__ __forceinline__ void async_copy16(const _Float16* g, _Float16* l) {
#if HAVE_ASYNC_LDS
  __builtin_amdgcn_global_load_async_to_lds_b128(
      (as1_v4i*)(as1_void*)g, (as3_v4i*)(as3_void*)l, 0, 0);
#else
  *(half8*)l = *(const half8*)g;
#endif
}

// Wait until at most N of this wave's async LDS-copies remain in flight.
// Async loads complete in order, so N=4 means "previous tile has landed"
// while the 4 just-issued copies for the next tile keep streaming.
template <int N>
static __device__ __forceinline__ void async_wait() {
#if HAVE_ASYNC_LDS
#if __has_builtin(__builtin_amdgcn_s_wait_asynccnt)
  __builtin_amdgcn_s_wait_asynccnt(N);
#else
  asm volatile("s_wait_asynccnt %0" ::"i"(N) : "memory");
#endif
#endif
}

// Build a 16-half fragment from two aligned 16-byte chunks.
static __device__ __forceinline__ v16h ld2x8(const _Float16* p0, const _Float16* p1) {
  union { v16h v; half8 h[2]; } u;
  u.h[0] = *(const half8*)p0;
  u.h[1] = *(const half8*)p1;
  return u.v;
}

// ---------------------------------------------------------------------------
// Kernel 0a: x (f32) -> Xh (f16), flat vectorized.
// ---------------------------------------------------------------------------
__global__ __launch_bounds__(256)
void cvt_x_kernel(const float* __restrict__ X, _Float16* __restrict__ Xh) {
  const size_t i = (size_t)blockIdx.x * blockDim.x + threadIdx.x;  // float4 index
  float4 xv = *(const float4*)&X[i * 4];
  half4 hx = { (_Float16)xv.x, (_Float16)xv.y, (_Float16)xv.z, (_Float16)xv.w };
  *(half4*)&Xh[i * 4] = hx;
}

// ---------------------------------------------------------------------------
// Kernel 0b: W[k][n] (f32, 512x512) -> WT[n][k] (f16), LDS-tiled transpose.
// blockIdx.z selects which of the 4 weight matrices.
// ---------------------------------------------------------------------------
__global__ __launch_bounds__(128)
void transpose_w_kernel(const float* __restrict__ Wq, const float* __restrict__ Wk,
                        const float* __restrict__ Wv, const float* __restrict__ Wo,
                        _Float16* __restrict__ WTq, _Float16* __restrict__ WTk,
                        _Float16* __restrict__ WTv, _Float16* __restrict__ WTo) {
  const int z = blockIdx.z;
  const float* W = (z == 0) ? Wq : (z == 1) ? Wk : (z == 2) ? Wv : Wo;
  _Float16* WT   = (z == 0) ? WTq : (z == 1) ? WTk : (z == 2) ? WTv : WTo;

  __shared__ _Float16 T[32 * 33];
  const int k0 = blockIdx.x * 32;
  const int n0 = blockIdx.y * 32;
  const int tid = threadIdx.x;

#pragma unroll
  for (int it = 0; it < 8; ++it) {
    int lin = it * 128 + tid;            // 0..1023
    int kk = lin >> 5, nn = lin & 31;    // coalesced read along n
    T[nn * 33 + kk] = (_Float16)W[(size_t)(k0 + kk) * DDIM + n0 + nn];
  }
  __syncthreads();
#pragma unroll
  for (int it = 0; it < 8; ++it) {
    int lin = it * 128 + tid;
    int nn = lin >> 5, kk = lin & 31;    // coalesced write along k
    WT[(size_t)(n0 + nn) * DDIM + k0 + kk] = T[nn * 33 + kk];
  }
}

// ---------------------------------------------------------------------------
// Kernel 1: fused QKV projection.  C[8192,512] = Xh @ W* + b*  (per z slice)
// A = Xh (f16 row-major), B = WT* (f16, [n][k]).
// Double-buffered async-DMA LDS staging overlapped with WMMA.
// z=0 -> Qh [B,H,N,HD] (pre-scaled), z=1 -> Kh [B,H,N,HD], z=2 -> Vt [B,H,HD,N]
// ---------------------------------------------------------------------------
__global__ __launch_bounds__(128)
void qkv_gemm_kernel(const _Float16* __restrict__ Xh,
                     const _Float16* __restrict__ WTq, const float* __restrict__ Bq,
                     const _Float16* __restrict__ WTk, const float* __restrict__ Bk,
                     const _Float16* __restrict__ WTv, const float* __restrict__ Bv,
                     _Float16* __restrict__ Qh, _Float16* __restrict__ Kh,
                     _Float16* __restrict__ Vt) {
  const int z = blockIdx.z;
  const _Float16* WT = (z == 0) ? WTq : ((z == 1) ? WTk : WTv);
  const float*    Bi = (z == 0) ? Bq : ((z == 1) ? Bk : Bv);

  __shared__ _Float16 As[2][64 * 40];  // Xh tile [row][k], stride 40 halves
  __shared__ _Float16 Bs[2][64 * 40];  // WT tile [n][k],  stride 40 halves

  const int tid  = threadIdx.x;
  const int lane = tid & 31;
  const int wv   = tid >> 5;   // wave 0..3
  const int l16  = lane & 15;
  const int hi   = lane >> 4;  // 0/1

  const int row0 = blockIdx.x * 64;
  const int col0 = blockIdx.y * 64;

  v8f acc[4] = {};

  // Stage one 64x32 tile pair (4 x 16B DMA chunks per thread).
  auto stage = [&](int k0, int buf) {
#pragma unroll
    for (int it = 0; it < 2; ++it) {
      int i = tid + 128 * it;           // 0..255
      int r = i >> 2, c = i & 3;        // row, 16B chunk within 32-half row
      async_copy16(&Xh[(size_t)(row0 + r) * DDIM + k0 + c * 8], &As[buf][r * 40 + c * 8]);
      async_copy16(&WT[(size_t)(col0 + r) * DDIM + k0 + c * 8], &Bs[buf][r * 40 + c * 8]);
    }
  };

  stage(0, 0);  // prologue

#pragma unroll
  for (int step = 0; step < KSTEPS; ++step) {
    const int buf = step & 1;
    if (step + 1 < KSTEPS) {
      stage((step + 1) * 32, buf ^ 1);  // prefetch next tile into other buffer
      async_wait<4>();                  // wait for *current* tile only
    } else {
      async_wait<0>();
    }
    __syncthreads();

    const int ar  = wv * 16 + l16;
    const int kb8 = hi * 8;
    v16h a = ld2x8(&As[buf][ar * 40 + kb8], &As[buf][ar * 40 + 16 + kb8]);
#pragma unroll
    for (int nt = 0; nt < 4; ++nt) {
      const int n = nt * 16 + l16;
      v16h b = ld2x8(&Bs[buf][n * 40 + hi * 8], &Bs[buf][n * 40 + 16 + hi * 8]);
      acc[nt] = __builtin_amdgcn_wmma_f32_16x16x32_f16(
          false, a, false, b, (short)0, acc[nt], false, false);
    }
    __syncthreads();  // all reads of buf done before it is DMA-overwritten
  }

  // Epilogue: uniform z branch hoisted outside the store loops.
  if (z == 2) {
#pragma unroll
    for (int nt = 0; nt < 4; ++nt) {
      const int col = col0 + nt * 16 + l16;
      const float bv = Bi[col];
      const int h = col >> 6, d = col & 63;
#pragma unroll
      for (int r = 0; r < 8; ++r) {
        const int R  = row0 + wv * 16 + r + hi * 8;
        const int bb = R >> 10, n_ = R & 1023;
        Vt[((size_t)(bb * HH + h) * HD + d) * NN + n_] = (_Float16)(acc[nt][r] + bv);
      }
    }
  } else {
    _Float16* dst = (z == 0) ? Qh : Kh;
    const float sc = (z == 0) ? ATT_SCALE : 1.0f;
#pragma unroll
    for (int nt = 0; nt < 4; ++nt) {
      const int col = col0 + nt * 16 + l16;
      const float bv = Bi[col];
      const int h = col >> 6, d = col & 63;
#pragma unroll
      for (int r = 0; r < 8; ++r) {
        const int R  = row0 + wv * 16 + r + hi * 8;
        const int bb = R >> 10, n_ = R & 1023;
        dst[((size_t)(bb * HH + h) * NN + n_) * HD + d] = (_Float16)((acc[nt][r] + bv) * sc);
      }
    }
  }
}

// ---------------------------------------------------------------------------
// Kernel 2: flash-style masked attention, one (b,h) column of 64 query rows
// per workgroup, 16 query rows per wave.  Oh layout: [B, N, H*HD] f16.
// ---------------------------------------------------------------------------
__global__ __launch_bounds__(128)
void attn_kernel(const _Float16* __restrict__ Qh,
                 const _Float16* __restrict__ Kh,
                 const _Float16* __restrict__ Vt,
                 const unsigned char* __restrict__ mask,  // jax bool = 1 byte
                 _Float16* __restrict__ Oh) {
  __shared__ _Float16 P[4][16 * 32];  // per-wave prob tile (C-layout -> A-layout)

  const int tid  = threadIdx.x;
  const int lane = tid & 31;
  const int wv   = tid >> 5;
  const int l16  = lane & 15;
  const int hi   = lane >> 4;

  const int bh = blockIdx.y;          // 0..63
  const int b  = bh >> 3;
  const int h  = bh & 7;
  const int q0 = blockIdx.x * 64 + wv * 16;

  const _Float16* Qp = Qh + (size_t)bh * NN * HD;
  const _Float16* Kp = Kh + (size_t)bh * NN * HD;
  const _Float16* Vp = Vt + (size_t)bh * HD * NN;
  const unsigned char* mp = mask + (size_t)b * NN;

  // Persistent Q A-fragments for this wave's 16 rows (d 0..31 and 32..63)
  const int qrow = q0 + l16;
  const int kb8  = hi * 8;
  v16h qa0 = ld2x8(&Qp[qrow * HD + kb8],      &Qp[qrow * HD + 16 + kb8]);
  v16h qa1 = ld2x8(&Qp[qrow * HD + 32 + kb8], &Qp[qrow * HD + 48 + kb8]);

  v8f o[4] = {};
  float m[8], l[8];
#pragma unroll
  for (int r = 0; r < 8; ++r) { m[r] = MASKNEG; l[r] = 0.f; }

  _Float16* Pw = &P[wv][0];

  for (int j = 0; j < NN / 32; ++j) {
    const int key0 = j * 32;
    v8f s[2];
#pragma unroll
    for (int t = 0; t < 2; ++t) {
      const int krow = key0 + t * 16 + l16;        // B-frag: lane%16 = key column
      v16h b0 = ld2x8(&Kp[krow * HD + hi * 8],      &Kp[krow * HD + 16 + hi * 8]);
      v16h b1 = ld2x8(&Kp[krow * HD + 32 + hi * 8], &Kp[krow * HD + 48 + hi * 8]);
      v8f st = {};
      st = __builtin_amdgcn_wmma_f32_16x16x32_f16(false, qa0, false, b0, (short)0, st, false, false);
      st = __builtin_amdgcn_wmma_f32_16x16x32_f16(false, qa1, false, b1, (short)0, st, false, false);
      const bool ok = mp[key0 + t * 16 + l16] != 0; // key-padding mask, branchless
#pragma unroll
      for (int r = 0; r < 8; ++r) st[r] = ok ? st[r] : MASKNEG;
      s[t] = st;
    }

    // Online softmax: row stats live across the 16-lane C-layout groups.
#pragma unroll
    for (int r = 0; r < 8; ++r) {
      float v = fmaxf(s[0][r], s[1][r]);
      v = fmaxf(v, __shfl_xor(v, 1));
      v = fmaxf(v, __shfl_xor(v, 2));
      v = fmaxf(v, __shfl_xor(v, 4));
      v = fmaxf(v, __shfl_xor(v, 8));
      const float mn = fmaxf(m[r], v);
      const float c  = __expf(m[r] - mn);
      const float p0 = __expf(s[0][r] - mn);
      const float p1 = __expf(s[1][r] - mn);
      float rs = p0 + p1;
      rs += __shfl_xor(rs, 1);
      rs += __shfl_xor(rs, 2);
      rs += __shfl_xor(rs, 4);
      rs += __shfl_xor(rs, 8);
      l[r] = l[r] * c + rs;
      m[r] = mn;
      o[0][r] *= c; o[1][r] *= c; o[2][r] *= c; o[3][r] *= c;
      s[0][r] = p0; s[1][r] = p1;
    }

    // C-layout -> row-major LDS -> A-layout fragment (wave-private region).
#pragma unroll
    for (int r = 0; r < 8; ++r) {
      const int prow = r + hi * 8;
      Pw[prow * 32 + l16]      = (_Float16)s[0][r];
      Pw[prow * 32 + 16 + l16] = (_Float16)s[1][r];
    }
    asm volatile("s_wait_dscnt 0" ::: "memory");  // CDNA5 split DS counter wait
    v16h pa = ld2x8(&Pw[l16 * 32 + kb8], &Pw[l16 * 32 + 16 + kb8]);

    // P(16x32) @ V(32x64): Vt rows give contiguous key runs per lane.
#pragma unroll
    for (int nt = 0; nt < 4; ++nt) {
      const _Float16* vrow = &Vp[(size_t)(nt * 16 + l16) * NN + key0 + hi * 8];
      v16h vb = ld2x8(vrow, vrow + 16);
      o[nt] = __builtin_amdgcn_wmma_f32_16x16x32_f16(
          false, pa, false, vb, (short)0, o[nt], false, false);
    }
  }

  // Normalize and emit f16 [B, N, H*HD] for the output projection.
  float inv[8];
#pragma unroll
  for (int r = 0; r < 8; ++r) inv[r] = 1.f / l[r];
#pragma unroll
  for (int nt = 0; nt < 4; ++nt) {
#pragma unroll
    for (int r = 0; r < 8; ++r) {
      const int row = q0 + r + hi * 8;
      const int d   = nt * 16 + l16;
      Oh[(size_t)(b * NN + row) * DDIM + h * HD + d] = (_Float16)(o[nt][r] * inv[r]);
    }
  }
}

// ---------------------------------------------------------------------------
// Kernel 3: output projection.  Y[8192,512] = Oh(f16) @ Wo + bo  (f32 out)
// A = Oh (f16 row-major), B = WTo (f16 [n][k]).  Double-buffered async DMA.
// ---------------------------------------------------------------------------
__global__ __launch_bounds__(128)
void out_gemm_kernel(const _Float16* __restrict__ Oh,
                     const _Float16* __restrict__ WTo,
                     const float* __restrict__ bo,
                     float* __restrict__ Y) {
  __shared__ _Float16 As[2][64 * 40];
  __shared__ _Float16 Bs[2][64 * 40];

  const int tid  = threadIdx.x;
  const int lane = tid & 31;
  const int wv   = tid >> 5;
  const int l16  = lane & 15;
  const int hi   = lane >> 4;

  const int row0 = blockIdx.x * 64;
  const int col0 = blockIdx.y * 64;

  v8f acc[4] = {};

  auto stage = [&](int k0, int buf) {
#pragma unroll
    for (int it = 0; it < 2; ++it) {
      int i = tid + 128 * it;
      int r = i >> 2, c = i & 3;
      async_copy16(&Oh[(size_t)(row0 + r) * DDIM + k0 + c * 8],  &As[buf][r * 40 + c * 8]);
      async_copy16(&WTo[(size_t)(col0 + r) * DDIM + k0 + c * 8], &Bs[buf][r * 40 + c * 8]);
    }
  };

  stage(0, 0);

#pragma unroll
  for (int step = 0; step < KSTEPS; ++step) {
    const int buf = step & 1;
    if (step + 1 < KSTEPS) {
      stage((step + 1) * 32, buf ^ 1);
      async_wait<4>();
    } else {
      async_wait<0>();
    }
    __syncthreads();

    const int ar  = wv * 16 + l16;
    const int kb8 = hi * 8;
    v16h a = ld2x8(&As[buf][ar * 40 + kb8], &As[buf][ar * 40 + 16 + kb8]);
#pragma unroll
    for (int nt = 0; nt < 4; ++nt) {
      const int n = nt * 16 + l16;
      v16h b = ld2x8(&Bs[buf][n * 40 + hi * 8], &Bs[buf][n * 40 + 16 + hi * 8]);
      acc[nt] = __builtin_amdgcn_wmma_f32_16x16x32_f16(
          false, a, false, b, (short)0, acc[nt], false, false);
    }
    __syncthreads();
  }

#pragma unroll
  for (int nt = 0; nt < 4; ++nt) {
    const int col = col0 + nt * 16 + l16;
    const float bv = bo[col];
#pragma unroll
    for (int r = 0; r < 8; ++r) {
      const int R = row0 + wv * 16 + r + hi * 8;
      Y[(size_t)R * DDIM + col] = acc[nt][r] + bv;
    }
  }
}

// ---------------------------------------------------------------------------
extern "C" void kernel_launch(void* const* d_in, const int* in_sizes, int n_in,
                              void* d_out, int out_size, void* d_ws, size_t ws_size,
                              hipStream_t stream) {
  const float* x  = (const float*)d_in[0];
  const unsigned char* mask = (const unsigned char*)d_in[1];  // jax bool layout
  const float* wq = (const float*)d_in[2];
  const float* bq = (const float*)d_in[3];
  const float* wk = (const float*)d_in[4];
  const float* bk = (const float*)d_in[5];
  const float* wv = (const float*)d_in[6];
  const float* bv = (const float*)d_in[7];
  const float* wo = (const float*)d_in[8];
  const float* bo = (const float*)d_in[9];
  float* out = (float*)d_out;

  // Workspace carve-up (f16 halves):
  //   Qh,Kh,Vt,Oh : 4 x B*H*N*HD = 4 x 4Mi halves (8 MiB each)
  //   Xh          : M*D          = 4Mi halves (8 MiB)
  //   WTq..WTo    : 4 x D*D      = 4 x 256Ki halves (512 KiB each)
  const size_t TSZ = (size_t)BB * HH * NN * HD;   // 4Mi
  const size_t WSZ = (size_t)DDIM * DDIM;         // 256Ki
  _Float16* Qh  = (_Float16*)d_ws;
  _Float16* Kh  = Qh + TSZ;
  _Float16* Vt  = Kh + TSZ;
  _Float16* Oh  = Vt + TSZ;
  _Float16* Xh  = Oh + TSZ;
  _Float16* WTq = Xh + TSZ;
  _Float16* WTk = WTq + WSZ;
  _Float16* WTv = WTk + WSZ;
  _Float16* WTo = WTv + WSZ;

  cvt_x_kernel<<<dim3((MM * DDIM) / (4 * 256)), 256, 0, stream>>>(x, Xh);
  transpose_w_kernel<<<dim3(DDIM / 32, DDIM / 32, 4), 128, 0, stream>>>(
      wq, wk, wv, wo, WTq, WTk, WTv, WTo);
  qkv_gemm_kernel<<<dim3(MM / 64, DDIM / 64, 3), 128, 0, stream>>>(
      Xh, WTq, bq, WTk, bk, WTv, bv, Qh, Kh, Vt);
  attn_kernel<<<dim3(NN / 64, BB * HH), 128, 0, stream>>>(Qh, Kh, Vt, mask, Oh);
  out_gemm_kernel<<<dim3(MM / 64, DDIM / 64), 128, 0, stream>>>(Oh, WTo, bo, out);
}
